// LSTMTagger_18889266168398
// MI455X (gfx1250) — 16-bit and 8-bit matrix operand order repaired and checked by reading, compiled, not run on hardware
//
#include <hip/hip_runtime.h>
#include <hip/hip_fp16.h>
#include <cstdint>

typedef int      v8i  __attribute__((ext_vector_type(8)));
typedef float    v8f  __attribute__((ext_vector_type(8)));
typedef _Float16 v16h __attribute__((ext_vector_type(16)));

#define LSEQ 2048
#define EDIM 512
#define HDIM 512
#define TDIM 1024
#define G4H  2048

__global__ void k_prep_x(const int* __restrict__ sent, const float* __restrict__ emb,
                         int8_t* __restrict__ x8) {
    int idx = blockIdx.x * blockDim.x + threadIdx.x;
    if (idx >= LSEQ * EDIM) return;
    int l = idx >> 9, e = idx & 511;
    long long row = (long long)sent[l];
    float v = emb[row * (long long)EDIM + e];
    x8[idx] = (v >= 0.f) ? (int8_t)1 : (int8_t)-1;
}

__global__ void k_sign8(const float* __restrict__ w, int8_t* __restrict__ o, int n) {
    int idx = blockIdx.x * blockDim.x + threadIdx.x;
    if (idx < n) o[idx] = (w[idx] >= 0.f) ? (int8_t)1 : (int8_t)-1;
}

__global__ void k_pack_whh(const float* __restrict__ whh, uint32_t* __restrict__ p) {
    int idx = blockIdx.x * blockDim.x + threadIdx.x;
    if (idx >= G4H * 16) return;
    int row = idx >> 4, w = idx & 15;
    const float* src = whh + row * HDIM + w * 32;
    uint32_t bits = 0u;
    #pragma unroll
    for (int b = 0; b < 32; ++b) bits |= (src[b] >= 0.f ? 1u : 0u) << b;
    p[idx] = bits;
}

__global__ void k_tohalf(const float* __restrict__ w, _Float16* __restrict__ o, int n) {
    int idx = blockIdx.x * blockDim.x + threadIdx.x;
    if (idx < n) o[idx] = (_Float16)w[idx];
}

__global__ void __launch_bounds__(256) k_gemm_ih(const int8_t* __restrict__ X,
                                                 const int8_t* __restrict__ W,
                                                 const float* __restrict__ bih,
                                                 const float* __restrict__ bhh,
                                                 float* __restrict__ xg) {
    int wave = (blockIdx.x << 3) + (threadIdx.x >> 5);
    int lane = threadIdx.x & 31;
    int mt = wave >> 5;
    int ng = wave & 31;
    int m0 = mt << 4, n0 = ng << 6;
    int lrow = lane & 15, hi = lane >> 4;

    v8i acc[4] = {{}, {}, {}, {}};
    const int8_t* arow = X + (m0 + lrow) * EDIM;
    const int8_t* brow = W + (n0 + lrow) * EDIM;
    for (int k0 = 0; k0 < EDIM; k0 += 64) {
        const uint2* ap = (const uint2*)(arow + k0 + (hi ? 16 : 0));
        uint2 a0 = ap[0], a1 = ap[1], a2 = ap[4], a3 = ap[5];
        v8i A;
        A[0]=(int)a0.x; A[1]=(int)a0.y; A[2]=(int)a1.x; A[3]=(int)a1.y;
        A[4]=(int)a2.x; A[5]=(int)a2.y; A[6]=(int)a3.x; A[7]=(int)a3.y;
        #pragma unroll
        for (int q = 0; q < 4; ++q) {
            const uint4* bp = (const uint4*)(brow + q * 16 * EDIM + k0 + (hi << 4));
            uint4 b0 = bp[0], b1 = bp[2];
            v8i B;
            B[0]=(int)b0.x; B[1]=(int)b0.y; B[2]=(int)b0.z; B[3]=(int)b0.w;
            B[4]=(int)b1.x; B[5]=(int)b1.y; B[6]=(int)b1.z; B[7]=(int)b1.w;
            acc[q] = __builtin_amdgcn_wmma_i32_16x16x64_iu8(true, A, true, B, acc[q],
                                                            false, false);
        }
    }
    #pragma unroll
    for (int q = 0; q < 4; ++q) {
        int col = n0 + (q << 4) + lrow;
        float bias = bih[col] + bhh[col];
        #pragma unroll
        for (int r = 0; r < 8; ++r) {
            int row = m0 + r + (hi << 3);
            xg[row * G4H + col] = (float)acc[q][r] + bias;
        }
    }
}

__global__ void __launch_bounds__(512) k_lstm(const float* __restrict__ xg,
                                              const uint32_t* __restrict__ Wp,
                                              _Float16* __restrict__ Hq) {
    __shared__ uint32_t hb[2][16];
    int t = threadIdx.x;
    uint32_t wi[16], wf[16], wg[16], wo[16];
    #pragma unroll
    for (int w = 0; w < 16; ++w) {
        wi[w] = Wp[(t         ) * 16 + w];
        wf[w] = Wp[(t +  512) * 16 + w];
        wg[w] = Wp[(t + 1024) * 16 + w];
        wo[w] = Wp[(t + 1536) * 16 + w];
    }
    if (t < 16) hb[0][t] = 0xffffffffu;
    float c = 0.f;
    float xgi = xg[t], xgf = xg[t + 512], xgg = xg[t + 1024], xgo = xg[t + 1536];
    __syncthreads();

    for (int step = 0; step < LSEQ; ++step) {
        const uint32_t* hr = hb[step & 1];
        int pi = 0, pf = 0, pg = 0, po = 0;
        #pragma unroll
        for (int w = 0; w < 16; ++w) {
            uint32_t hx = hr[w];
            pi += __popc(hx ^ wi[w]);
            pf += __popc(hx ^ wf[w]);
            pg += __popc(hx ^ wg[w]);
            po += __popc(hx ^ wo[w]);
        }
        float gi_v = xgi + (float)(512 - 2 * pi);
        float gf_v = xgf + (float)(512 - 2 * pf);
        float gg_v = xgg + (float)(512 - 2 * pg);
        float go_v = xgo + (float)(512 - 2 * po);

        int ns = (step + 1 < LSEQ) ? (step + 1) : step;
        const float* nx = xg + ns * G4H;
        xgi = nx[t]; xgf = nx[t + 512]; xgg = nx[t + 1024]; xgo = nx[t + 1536];

        float ig = 1.f / (1.f + __expf(-gi_v));
        float fg = 1.f / (1.f + __expf(-gf_v));
        float gg = tanhf(gg_v);
        float og = 1.f / (1.f + __expf(-go_v));
        float cq = (c >= 0.f) ? 1.f : -1.f;
        c = fg * cq + ig * gg;
        float h = og * tanhf(c);
        bool bit = (h >= 0.f);
        Hq[(size_t)step * HDIM + t] = bit ? (_Float16)1.f : (_Float16)-1.f;
        unsigned int m = __builtin_amdgcn_ballot_w32(bit);
        if ((t & 31) == 0) hb[(step + 1) & 1][t >> 5] = m;
        __syncthreads();
    }
}

__global__ void __launch_bounds__(256) k_gemm_out(const _Float16* __restrict__ Hq,
                                                  const _Float16* __restrict__ W,
                                                  const float* __restrict__ bout,
                                                  float* __restrict__ out) {
    int wave = (blockIdx.x << 3) + (threadIdx.x >> 5);
    int lane = threadIdx.x & 31;
    int mt = wave >> 4;
    int ng = wave & 15;
    int m0 = mt << 4, n0 = ng << 6;
    int lrow = lane & 15, hi = lane >> 4;

    v8f acc[4] = {{}, {}, {}, {}};
    const _Float16* arow = Hq + (m0 + lrow) * HDIM;
    const _Float16* brow = W  + (n0 + lrow) * HDIM;
    for (int k0 = 0; k0 < HDIM; k0 += 32) {
        union { uint4 u[2]; v16h v; } A;
        const uint4* ap = (const uint4*)(arow + k0 + (hi << 4));
        A.u[0] = ap[0]; A.u[1] = ap[1];
        #pragma unroll
        for (int q = 0; q < 4; ++q) {
            union { uint4 u[2]; v16h v; } B;
            const uint4* bp = (const uint4*)(brow + q * 16 * HDIM + k0 + (hi << 4));
            B.u[0] = bp[0]; B.u[1] = bp[1];
            acc[q] = __builtin_amdgcn_wmma_f32_16x16x32_f16(false, A.v, false, B.v,
                                                            (short)0, acc[q],
                                                            false, false);
        }
    }
    #pragma unroll
    for (int q = 0; q < 4; ++q) {
        int col = n0 + (q << 4) + lrow;
        float bias = bout[col];
        #pragma unroll
        for (int r = 0; r < 8; ++r) {
            int row = m0 + r + (hi << 3);
            out[row * TDIM + col] = acc[q][r] + bias;
        }
    }
}

__global__ void __launch_bounds__(256) k_logsoftmax(float* __restrict__ out) {
    __shared__ float red[256];
    float* p = out + (size_t)blockIdx.x * TDIM;
    int t = threadIdx.x;
    float m = -3.4e38f;
    for (int i = t; i < TDIM; i += 256) m = fmaxf(m, p[i]);
    red[t] = m; __syncthreads();
    for (int s = 128; s > 0; s >>= 1) { if (t < s) red[t] = fmaxf(red[t], red[t + s]); __syncthreads(); }
    m = red[0]; __syncthreads();
    float sum = 0.f;
    for (int i = t; i < TDIM; i += 256) sum += __expf(p[i] - m);
    red[t] = sum; __syncthreads();
    for (int s = 128; s > 0; s >>= 1) { if (t < s) red[t] += red[t + s]; __syncthreads(); }
    float lse = m + __logf(red[0]);
    for (int i = t; i < TDIM; i += 256) p[i] = p[i] - lse;
}

extern "C" void kernel_launch(void* const* d_in, const int* in_sizes, int n_in,
                              void* d_out, int out_size, void* d_ws, size_t ws_size,
                              hipStream_t stream) {
    const int*   sent = (const int*)d_in[0];
    const float* emb  = (const float*)d_in[1];
    const float* Wih  = (const float*)d_in[2];
    const float* Whh  = (const float*)d_in[3];
    const float* bih  = (const float*)d_in[4];
    const float* bhh  = (const float*)d_in[5];
    const float* Wout = (const float*)d_in[6];
    const float* bout = (const float*)d_in[7];
    float* out = (float*)d_out;

    char* ws = (char*)d_ws;
    int8_t*   X8  = (int8_t*)ws;
    int8_t*   W8  = (int8_t*)(ws + (1u << 20));
    uint32_t* Wp  = (uint32_t*)(ws + (2u << 20));
    _Float16* WoH = (_Float16*)(ws + (2u << 20) + (1u << 17));
    _Float16* Hq  = (_Float16*)(ws + (3u << 20) + (1u << 17));
    float*    XG  = (float*)(ws + (5u << 20) + (1u << 17));

    k_prep_x  <<<(LSEQ * EDIM + 255) / 256, 256, 0, stream>>>(sent, emb, X8);
    k_sign8   <<<(G4H * EDIM + 255) / 256, 256, 0, stream>>>(Wih, W8, G4H * EDIM);
    k_pack_whh<<<(G4H * 16 + 255) / 256, 256, 0, stream>>>(Whh, Wp);
    k_tohalf  <<<(TDIM * HDIM + 255) / 256, 256, 0, stream>>>(Wout, WoH, TDIM * HDIM);

    k_gemm_ih <<<512, 256, 0, stream>>>(X8, W8, bih, bhh, XG);
    k_lstm    <<<1, 512, 0, stream>>>(XG, Wp, Hq);
    k_gemm_out<<<256, 256, 0, stream>>>(Hq, WoH, bout, out);
    k_logsoftmax<<<LSEQ, 256, 0, stream>>>(out);
}
